// SDFNetwork_35785667510293
// MI455X (gfx1250) — matrix operand order repaired and checked by reading, not compiled, not run on hardware
//
#include <hip/hip_runtime.h>
#include <cmath>

typedef __attribute__((ext_vector_type(16))) _Float16 v16h;
typedef __attribute__((ext_vector_type(4)))  _Float16 v4h;
typedef __attribute__((ext_vector_type(2)))  _Float16 v2h;
typedef __attribute__((ext_vector_type(8)))  float    v8f;

struct LevelMeta {
  float    scale;    // BASE_RES * GROWTH^l - 1
  unsigned stride;   // res + 1 (dense addressing)
  unsigned mask;     // params - 1 (hash levels; params is a power of two here)
  unsigned offset;   // cumulative table offset (in feature rows)
  unsigned dense;    // 1 = dense indexing, 0 = hash
};
struct Metas { LevelMeta lvl[16]; };

#define WAVES_PER_BLOCK 8
#define ROWS_PER_WAVE   16
#define ROWS_PER_BLOCK  (WAVES_PER_BLOCK * ROWS_PER_WAVE)

// ---- A fragment: 16x32 f16, lane (half*16+m) holds row M=m.
// ISA layout: lanes 0-15: V0..V3 -> K=0..7, V4..V7 -> K=16..23
//             lanes 16-31: V0..V3 -> K=8..15, V4..V7 -> K=24..31
__device__ __forceinline__ v16h load_a_frag(const _Float16* hb, int kc, int half, int m) {
  v16h a;
  const _Float16* row = hb + m * 64 + kc * 32;
#pragma unroll
  for (int v = 0; v < 8; ++v) {
    int k = (v < 4) ? (8 * half + 2 * v) : (16 + 8 * half + 2 * (v - 4));
    v2h p = *(const v2h*)(row + k);   // ds_load_b32
    a[2 * v]     = p.x;
    a[2 * v + 1] = p.y;
  }
  return a;
}

// ---- B fragment: 32x16 f16, lanes 0-15 hold K=0..7,16..23 of column N=lane,
// lanes 16-31 hold K=8..15,24..31 of column N=lane-16 (matches A fragment order).
// B[k][n] = W[n][k] (out = h @ W^T), W is row-major f32 [64][64].
__device__ __forceinline__ v16h load_b_frag(const float* __restrict__ W, int n, int kc, int half) {
  v16h b;
  const float* wr = W + n * 64 + kc * 32 + half * 8;
#pragma unroll
  for (int v = 0; v < 8; ++v) {
    int o = (v < 4) ? (2 * v) : (2 * v + 8);
    b[2 * v]     = (_Float16)wr[o];
    b[2 * v + 1] = (_Float16)wr[o + 1];
  }
  return b;
}

// ---- One 64->64 layer with ReLU: hin (16x64 f16 LDS) -> hout (16x64 f16 LDS)
__device__ __forceinline__ void mlp_layer_relu(const _Float16* hin, _Float16* hout,
                                               const float* __restrict__ W,
                                               int half, int nIdx) {
  v16h a0 = load_a_frag(hin, 0, half, nIdx);
  v16h a1 = load_a_frag(hin, 1, half, nIdx);
#pragma unroll
  for (int t = 0; t < 4; ++t) {
    int n = t * 16 + nIdx;
    v16h b0 = load_b_frag(W, n, 0, half);
    v16h b1 = load_b_frag(W, n, 1, half);
    v8f acc = {0.f, 0.f, 0.f, 0.f, 0.f, 0.f, 0.f, 0.f};
    acc = __builtin_amdgcn_wmma_f32_16x16x32_f16(false, a0, false, b0, (short)0, acc, false, false);
    acc = __builtin_amdgcn_wmma_f32_16x16x32_f16(false, a1, false, b1, (short)0, acc, false, false);
    // D layout: VGPR r = element (M = 8*half + r, N = lane%16) of this tile
#pragma unroll
    for (int r = 0; r < 8; ++r) {
      float vv = fmaxf(acc[r], 0.0f);
      hout[(half * 8 + r) * 64 + n] = (_Float16)vv;
    }
  }
}

__global__ __launch_bounds__(256)
void fused_hashgrid_mlp(const float* __restrict__ x,
                        const float* __restrict__ table,
                        const float* __restrict__ W0,
                        const float* __restrict__ W1,
                        const float* __restrict__ W2,
                        float* __restrict__ out,
                        Metas metas, int npts)
{
  __shared__ _Float16 smem[WAVES_PER_BLOCK][2][ROWS_PER_WAVE * 64];

  const int tid  = threadIdx.x;
  const int wave = tid >> 5;
  const int lane = tid & 31;
  const int half = lane >> 4;     // 0: levels 0-7 / K-low half, 1: levels 8-15 / K-high half
  const int nIdx = lane & 15;     // row / column index within a 16-tile
  const int m0   = (blockIdx.x * WAVES_PER_BLOCK + wave) * ROWS_PER_WAVE;

  int prow = m0 + nIdx;
  if (prow > npts - 1) prow = npts - 1;   // clamp keeps EXEC full for WMMA

  const float px = x[3 * prow + 0];
  const float py = x[3 * prow + 1];
  const float pz = x[3 * prow + 2];

  _Float16* h0 = &smem[wave][0][0];
  _Float16* h1 = &smem[wave][1][0];

  // ---- Hashgrid encode: this lane does levels half*8 .. half*8+7 of point prow,
  // writing 4 f16 features per level into row nIdx of h0.
#pragma unroll
  for (int li = 0; li < 8; ++li) {
    LevelMeta lo = metas.lvl[li];
    LevelMeta hi = metas.lvl[li + 8];
    float    scale  = half ? hi.scale  : lo.scale;
    unsigned stride = half ? hi.stride : lo.stride;
    unsigned mask   = half ? hi.mask   : lo.mask;
    unsigned offs   = half ? hi.offset : lo.offset;
    unsigned dense  = half ? hi.dense  : lo.dense;

    float fx = px * scale + 0.5f;
    float fy = py * scale + 0.5f;
    float fz = pz * scale + 0.5f;
    float flx = floorf(fx), fly = floorf(fy), flz = floorf(fz);
    float frx = fx - flx,   fry = fy - fly,   frz = fz - flz;
    unsigned bx = (unsigned)(int)flx;
    unsigned by = (unsigned)(int)fly;
    unsigned bz = (unsigned)(int)flz;
    float wx0 = 1.0f - frx, wy0 = 1.0f - fry, wz0 = 1.0f - frz;

    float acc0 = 0.f, acc1 = 0.f, acc2 = 0.f, acc3 = 0.f;
#pragma unroll
    for (int c = 0; c < 8; ++c) {
      unsigned cx = bx + (c & 1);
      unsigned cy = by + ((c >> 1) & 1);
      unsigned cz = bz + ((c >> 2) & 1);
      unsigned id;
      if (dense) {
        id = cx + cy * stride + cz * stride * stride;
      } else {
        id = (cx ^ (cy * 2654435761u) ^ (cz * 805459861u)) & mask;
      }
      id += offs;
      float w = ((c & 1) ? frx : wx0)
              * (((c >> 1) & 1) ? fry : wy0)
              * (((c >> 2) & 1) ? frz : wz0);
      const float4 f = *((const float4*)table + id);   // global_load_b128 gather
      acc0 = fmaf(w, f.x, acc0);
      acc1 = fmaf(w, f.y, acc1);
      acc2 = fmaf(w, f.z, acc2);
      acc3 = fmaf(w, f.w, acc3);
    }
    int col = (half * 8 + li) * 4;
    v4h packed = { (_Float16)acc0, (_Float16)acc1, (_Float16)acc2, (_Float16)acc3 };
    *(v4h*)(h0 + nIdx * 64 + col) = packed;            // ds_store_b64
  }
  __syncthreads();

  // ---- MLP layers 0 and 1 (64->64, ReLU) via v_wmma_f32_16x16x32_f16
  mlp_layer_relu(h0, h1, W0, half, nIdx);
  __syncthreads();
  mlp_layer_relu(h1, h0, W1, half, nIdx);
  __syncthreads();

  // ---- Final 64->1 layer: per-row dot product, halves split K, shfl reduce.
  float s = 0.f;
  const _Float16* hr = h0 + nIdx * 64 + half * 32;
  const float* w2 = W2 + half * 32;
#pragma unroll
  for (int k = 0; k < 32; ++k) s = fmaf((float)hr[k], w2[k], s);
  s += __shfl_xor(s, 16, 32);
  if (half == 0 && (m0 + nIdx) < npts) out[m0 + nIdx] = s;
}

extern "C" void kernel_launch(void* const* d_in, const int* in_sizes, int n_in,
                              void* d_out, int out_size, void* d_ws, size_t ws_size,
                              hipStream_t stream) {
  (void)n_in; (void)out_size; (void)d_ws; (void)ws_size;
  const float* x     = (const float*)d_in[0];
  const float* table = (const float*)d_in[1];
  const float* W0    = (const float*)d_in[2];
  const float* W1    = (const float*)d_in[3];
  const float* W2    = (const float*)d_in[4];
  float* out = (float*)d_out;
  const int npts = in_sizes[0] / 3;

  // Level metadata, matching the reference _level_metas() exactly (double math).
  Metas metas;
  const double growth = exp((log(2048.0) - log(16.0)) / 15.0);
  unsigned long long offset = 0;
  for (int l = 0; l < 16; ++l) {
    double scale = 16.0 * pow(growth, (double)l) - 1.0;
    int res = (int)ceil(scale) + 1;
    unsigned long long side  = (unsigned long long)res + 1ull;
    unsigned long long dsize = side * side * side;
    unsigned long long params = dsize < (1ull << 22) ? dsize : (1ull << 22);
    params = ((params + 7ull) / 8ull) * 8ull;
    metas.lvl[l].scale  = (float)scale;
    metas.lvl[l].stride = (unsigned)side;
    metas.lvl[l].mask   = (unsigned)(params - 1ull);  // hash levels: params == 2^22 (pow2)
    metas.lvl[l].offset = (unsigned)offset;
    metas.lvl[l].dense  = (dsize <= params) ? 1u : 0u;
    offset += params;
  }

  const int nblocks = (npts + ROWS_PER_BLOCK - 1) / ROWS_PER_BLOCK;
  hipLaunchKernelGGL(fused_hashgrid_mlp, dim3(nblocks), dim3(256), 0, stream,
                     x, table, W0, W1, W2, out, metas, npts);
}
